// CGC_MULTI_LAYER_88785563943285
// MI455X (gfx1250) — matrix operand order repaired and checked by reading, not compiled, not run on hardware
//
#include <hip/hip_runtime.h>
#include <hip/hip_bf16.h>

// ---------------------------------------------------------------------------
// Problem constants (match reference)
// ---------------------------------------------------------------------------
#define BN      4096   // batch
#define DD      1024   // input dim
#define HH1     512    // hidden 1
#define HH2     256    // hidden 2
#define NEXP    12     // total experts (4 shared + 4 task0 + 4 task1)

typedef __bf16 bf16_t;
typedef __attribute__((ext_vector_type(4)))  float    v4f;
typedef __attribute__((ext_vector_type(8)))  __bf16   v8bf;
typedef __attribute__((ext_vector_type(16))) __bf16   v16bf;
typedef __attribute__((ext_vector_type(8)))  float    v8f;
typedef __attribute__((ext_vector_type(4)))  unsigned u32x4;
typedef __attribute__((ext_vector_type(8)))  int      i32x8;
typedef __attribute__((ext_vector_type(4)))  int      i32x4;

#if __has_builtin(__builtin_amdgcn_tensor_load_to_lds) && \
    __has_builtin(__builtin_amdgcn_s_wait_tensorcnt)
#define HAVE_TDM 1
#else
#define HAVE_TDM 0
#endif

// ---------------------------------------------------------------------------
// TDM: one-instruction 2D tile DMA, global (bf16, row stride `stride_elems`)
// -> LDS (dense [tile_n][tile_k]).  Descriptor per CDNA5 ISA ch.8.
// 6-arg builtin (clang-23 / therock-10.0 lane).
// ---------------------------------------------------------------------------
#if HAVE_TDM
__device__ __forceinline__ void tdm_load_2d(unsigned lds_off, const bf16_t* gsrc,
                                            unsigned tile_k, unsigned tile_n,
                                            unsigned tensor_k, unsigned tensor_n,
                                            unsigned stride_elems) {
    unsigned long long ga = (unsigned long long)(uintptr_t)gsrc;
    u32x4 g0;
    g0[0] = 1u;                                             // count=1 (valid)
    g0[1] = lds_off;                                        // lds_addr (bytes)
    g0[2] = (unsigned)(ga & 0xFFFFFFFFu);                   // global_addr[31:0]
    g0[3] = (unsigned)((ga >> 32) & 0x01FFFFFFu) | (2u << 30); // addr[56:32] | type=2
    i32x8 g1;
    g1[0] = (int)(1u << 16);                                // data_size=1 -> 2 bytes
    g1[1] = (int)((tensor_k & 0xFFFFu) << 16);              // tensor_dim0[15:0]
    g1[2] = (int)(((tensor_k >> 16) & 0xFFFFu) |            // tensor_dim0[31:16]
                  ((tensor_n & 0xFFFFu) << 16));            // tensor_dim1[15:0]
    g1[3] = (int)(((tensor_n >> 16) & 0xFFFFu) |            // tensor_dim1[31:16]
                  ((tile_k & 0xFFFFu) << 16));              // tile_dim0
    g1[4] = (int)(tile_n & 0xFFFFu);                        // tile_dim1 (tile_dim2=0)
    g1[5] = (int)stride_elems;                              // tensor_dim0_stride[31:0]
    g1[6] = 0;
    g1[7] = 0;
    i32x4 gz4 = {0, 0, 0, 0};
    i32x8 gz8 = {0, 0, 0, 0, 0, 0, 0, 0};
    __builtin_amdgcn_tensor_load_to_lds(g0, g1, gz4, gz4, gz8, 0);
}
#endif

// ---------------------------------------------------------------------------
// WMMA fragment helpers (CDNA5 wave32 layouts, 16x16x32 bf16)
// ---------------------------------------------------------------------------
__device__ __forceinline__ v16bf frag_a_lds(const bf16_t* base, int stride, int lane) {
    const int m  = lane & 15;
    const int ko = (lane >> 4) << 3;
    const bf16_t* p = base + m * stride + ko;
    v8bf lo = *(const v8bf*)(p);
    v8bf hi = *(const v8bf*)(p + 16);
    return __builtin_shufflevector(lo, hi, 0,1,2,3,4,5,6,7,8,9,10,11,12,13,14,15);
}

__device__ __forceinline__ v16bf frag_b_lds(const bf16_t* base, int stride, int lane) {
    const int n  = lane & 15;
    const int kb = (lane >> 4) << 3;
    const bf16_t* p = base + n * stride + kb;
    v8bf lo = *(const v8bf*)(p);
    v8bf hi = *(const v8bf*)(p + 16);
    return __builtin_shufflevector(lo, hi, 0,1,2,3,4,5,6,7,8,9,10,11,12,13,14,15);
}

__device__ __forceinline__ v8f wmma_bf16(v16bf a, v16bf b, v8f c) {
    return __builtin_amdgcn_wmma_f32_16x16x32_bf16(
        false, a, false, b, (short)0, c, false, false);
}

// ---------------------------------------------------------------------------
// Pre-conversion kernels
// ---------------------------------------------------------------------------
__global__ __launch_bounds__(256)
void cgc_convert_x_kernel(const float* __restrict__ xs, const float* __restrict__ x0,
                          const float* __restrict__ x1, bf16_t* __restrict__ Xbf) {
    const size_t i4 = ((size_t)blockIdx.x * 256 + threadIdx.x) * 4;
    const size_t per = (size_t)BN * DD;
    const int sel = (int)(i4 / per);
    const size_t off = i4 - (size_t)sel * per;
    const float* src = (sel == 0) ? xs : (sel == 1) ? x0 : x1;
    v4f v = *(const v4f*)(src + off);
    bf16_t* dst = Xbf + (size_t)sel * per + off;
    dst[0] = (bf16_t)v[0]; dst[1] = (bf16_t)v[1];
    dst[2] = (bf16_t)v[2]; dst[3] = (bf16_t)v[3];
}

__global__ __launch_bounds__(256)
void cgc_convert_wt_kernel(const float* __restrict__ W, bf16_t* __restrict__ Wt,
                           int K, int N) {
    __shared__ float tile[32][33];
    const int e  = blockIdx.z;
    const int n0 = blockIdx.x * 32;
    const int k0 = blockIdx.y * 32;
    const int tid = threadIdx.x;
    const float* We  = W  + (size_t)e * K * N;
    bf16_t*      Wte = Wt + (size_t)e * K * N;
    for (int i = tid; i < 1024; i += 256) {
        const int kk = i >> 5, nn = i & 31;
        tile[kk][nn] = We[(size_t)(k0 + kk) * N + (n0 + nn)];
    }
    __syncthreads();
    for (int i = tid; i < 1024; i += 256) {
        const int nn = i >> 5, kk = i & 31;
        Wte[(size_t)(n0 + nn) * K + (k0 + kk)] = (bf16_t)tile[kk][nn];
    }
}

// ---------------------------------------------------------------------------
// Kernel 1: gating softmaxes.  One block per batch row.
// ---------------------------------------------------------------------------
__global__ __launch_bounds__(256)
void cgc_gates_kernel(const float* __restrict__ x0, const float* __restrict__ x1,
                      const float* __restrict__ xs,
                      const float* __restrict__ Wg0, const float* __restrict__ Wg1,
                      const float* __restrict__ Wgs,
                      float* __restrict__ g0, float* __restrict__ g1,
                      float* __restrict__ gs) {
    __shared__ float xr[3][DD];
    __shared__ float part[28][8];
    __shared__ float logit[28];

    const int b   = blockIdx.x;
    const int tid = threadIdx.x;

    for (int i = tid; i < DD; i += 256) {
        xr[0][i] = x0[(size_t)b * DD + i];
        xr[1][i] = x1[(size_t)b * DD + i];
        xr[2][i] = xs[(size_t)b * DD + i];
    }
    __syncthreads();

    if (tid < 224) {
        const int c     = tid >> 3;
        const int chunk = tid & 7;
        const float* xrow; const float* Wg; int col, ncol;
        if (c < 8)       { xrow = xr[0]; Wg = Wg0; col = c;      ncol = 8;  }
        else if (c < 16) { xrow = xr[1]; Wg = Wg1; col = c - 8;  ncol = 8;  }
        else             { xrow = xr[2]; Wg = Wgs; col = c - 16; ncol = 12; }
        float s = 0.f;
        const int i0 = chunk * 128;
        for (int i = i0; i < i0 + 128; ++i)
            s += xrow[i] * Wg[(size_t)i * ncol + col];
        part[c][chunk] = s;
    }
    __syncthreads();

    if (tid < 28) {
        float s = 0.f;
        for (int j = 0; j < 8; ++j) s += part[tid][j];
        logit[tid] = s;
    }
    __syncthreads();

    if (tid < 3) {
        const int base = (tid == 0) ? 0 : (tid == 1) ? 8 : 16;
        const int n    = (tid == 2) ? 12 : 8;
        float mx = -1e30f;
        for (int j = 0; j < n; ++j) mx = fmaxf(mx, logit[base + j]);
        float ex[12]; float sum = 0.f;
        for (int j = 0; j < n; ++j) { ex[j] = __expf(logit[base + j] - mx); sum += ex[j]; }
        const float inv = 1.f / sum;
        float* out = (tid == 0) ? (g0 + (size_t)b * 8)
                   : (tid == 1) ? (g1 + (size_t)b * 8)
                   :              (gs + (size_t)b * 12);
        for (int j = 0; j < n; ++j) out[j] = ex[j] * inv;
    }
}

// ---------------------------------------------------------------------------
// Kernel 2: fused two-layer expert MLP; bf16 WMMA, f32 accumulation.
// grid = (B/32, 12); 256 threads = 8 wave32 (2 M-tiles x 4 N-groups).
// Double-buffered TDM staging: while waves compute chunk kc, the TDM engine
// streams chunk kc+1 into the alternate LDS buffer (wait tensorcnt<=2 keeps
// only the prefetch in flight; tensor ops retire in order per wave).
// LDS (dynamic, bytes):
//   [0,2K)      Xs[0]   32x32  bf16      [2K,4K)     Xs[1]
//   [4K,36K)    Wt[0]   512x32 bf16      [36K,68K)   Wt[1]
//   [68K,+33K)  Hs      32x520 bf16
// ---------------------------------------------------------------------------
#define HS_STRIDE  520
#define SMEM_XS0   0
#define SMEM_XS1   2048
#define SMEM_WT0   4096
#define SMEM_WT1   36864
#define SMEM_HS    69632
#define SMEM_TOTAL 102912

__global__ __launch_bounds__(256)
void cgc_expert_ffn_kernel(const bf16_t* __restrict__ Xbf /* [3][B][D] */,
                           const bf16_t* __restrict__ W1t /* [12][H1][D] */,
                           const float*  __restrict__ b1,
                           const bf16_t* __restrict__ W2t /* [12][H2][H1] */,
                           const float*  __restrict__ b2,
                           float* __restrict__ Eout /* [12][B][H2] */) {
    extern __shared__ char smem[];
    bf16_t* Hs  = (bf16_t*)(smem + SMEM_HS);
    bf16_t* Xs0 = (bf16_t*)(smem + SMEM_XS0);
    bf16_t* Xs1 = (bf16_t*)(smem + SMEM_XS1);
    bf16_t* Wt0 = (bf16_t*)(smem + SMEM_WT0);
    bf16_t* Wt1 = (bf16_t*)(smem + SMEM_WT1);

    const int e    = blockIdx.y;
    const int row0 = blockIdx.x * 32;
    const int tid  = threadIdx.x;
    const int lane = tid & 31;
    const int wave = tid >> 5;
    const int mt   = wave >> 2;
    const int ng   = wave & 3;

    const int sel = (e < 4) ? 0 : (e < 8) ? 1 : 2;          // xs / x0 / x1
    const bf16_t* Xe   = Xbf + (size_t)sel * BN * DD;
    const bf16_t* W1te = W1t + (size_t)e * HH1 * DD;
    const bf16_t* W2te = W2t + (size_t)e * HH2 * HH1;

    // ---------------- layer 1: h = relu(X @ W1 + b1) ----------------
    v8f acc[8];
#pragma unroll
    for (int t = 0; t < 8; ++t)
#pragma unroll
        for (int r = 0; r < 8; ++r) acc[t][r] = 0.f;

    const int NK1 = DD / 32;
#if HAVE_TDM
    if (wave == 0) {                                        // prologue: buffer 0
        tdm_load_2d(SMEM_XS0, Xe + (size_t)row0 * DD, 32, 32, DD, BN, DD);
        tdm_load_2d(SMEM_WT0, W1te, 32, HH1, DD, HH1, DD);
    }
#endif
    for (int kc = 0; kc < NK1; ++kc) {
        const int odd = kc & 1;
        bf16_t* Xcur = odd ? Xs1 : Xs0;                     // s_cselect, no mem lookup
        bf16_t* Wcur = odd ? Wt1 : Wt0;
        __syncthreads();   // prior compute on the other buffers finished everywhere
#if HAVE_TDM
        if (wave == 0) {
            if (kc + 1 < NK1) {
                const unsigned xsN = odd ? SMEM_XS0 : SMEM_XS1;
                const unsigned wtN = odd ? SMEM_WT0 : SMEM_WT1;
                tdm_load_2d(xsN, Xe + (size_t)row0 * DD + (kc + 1) * 32,
                            32, 32, DD, BN, DD);
                tdm_load_2d(wtN, W1te + (kc + 1) * 32,
                            32, HH1, DD, HH1, DD);
                __builtin_amdgcn_s_wait_tensorcnt(2);       // `cur` tiles landed
            } else {
                __builtin_amdgcn_s_wait_tensorcnt(0);
            }
        }
#else
        for (int i = tid; i < 32 * 4; i += 256) {
            const int m = i >> 2, s = i & 3;
            *(v8bf*)(Xcur + m * 32 + s * 8) =
                *(const v8bf*)(Xe + (size_t)(row0 + m) * DD + kc * 32 + s * 8);
        }
        for (int i = tid; i < HH1 * 4; i += 256) {
            const int n = i >> 2, s = i & 3;
            *(v8bf*)(Wcur + n * 32 + s * 8) =
                *(const v8bf*)(W1te + (size_t)n * DD + kc * 32 + s * 8);
        }
#endif
        __syncthreads();

        const v16bf a = frag_a_lds(Xcur + mt * 16 * 32, 32, lane);
        // one-deep B-fragment pipeline: load nt+1 while WMMA nt executes
        v16bf bcur = frag_b_lds(Wcur + (ng * 128) * 32, 32, lane);
#pragma unroll
        for (int nt = 0; nt < 8; ++nt) {
            v16bf bnext = (nt < 7)
                ? frag_b_lds(Wcur + (ng * 128 + (nt + 1) * 16) * 32, 32, lane)
                : bcur;
            acc[nt] = wmma_bf16(a, bcur, acc[nt]);
            bcur = bnext;
        }
    }

    // bias + relu -> hidden tile in LDS (bf16)
    {
        const float* b1e  = b1 + (size_t)e * HH1;
        const int    mrow = mt * 16 + ((lane >> 4) << 3);
#pragma unroll
        for (int nt = 0; nt < 8; ++nt) {
            const int   n    = ng * 128 + nt * 16 + (lane & 15);
            const float bias = b1e[n];
#pragma unroll
            for (int r = 0; r < 8; ++r)
                Hs[(mrow + r) * HS_STRIDE + n] = (bf16_t)fmaxf(acc[nt][r] + bias, 0.f);
        }
    }
    __syncthreads();

    // ---------------- layer 2: o = relu(h @ W2 + b2) ----------------
    v8f acc2[4];
#pragma unroll
    for (int t = 0; t < 4; ++t)
#pragma unroll
        for (int r = 0; r < 8; ++r) acc2[t][r] = 0.f;

    const int NK2 = HH1 / 32;
#if HAVE_TDM
    if (wave == 0) {                                        // prologue: buffer 0
        tdm_load_2d(SMEM_WT0, W2te, 32, HH2, HH1, HH2, HH1);
    }
#endif
    for (int kc = 0; kc < NK2; ++kc) {
        const int odd = kc & 1;
        bf16_t* Wcur = odd ? Wt1 : Wt0;
        __syncthreads();
#if HAVE_TDM
        if (wave == 0) {
            if (kc + 1 < NK2) {
                const unsigned wtN = odd ? SMEM_WT0 : SMEM_WT1;
                tdm_load_2d(wtN, W2te + (kc + 1) * 32,
                            32, HH2, HH1, HH2, HH1);
                __builtin_amdgcn_s_wait_tensorcnt(1);
            } else {
                __builtin_amdgcn_s_wait_tensorcnt(0);
            }
        }
#else
        for (int i = tid; i < HH2 * 4; i += 256) {
            const int n = i >> 2, s = i & 3;
            *(v8bf*)(Wcur + n * 32 + s * 8) =
                *(const v8bf*)(W2te + (size_t)n * HH1 + kc * 32 + s * 8);
        }
#endif
        __syncthreads();

        const v16bf a = frag_a_lds(Hs + mt * 16 * HS_STRIDE + kc * 32, HS_STRIDE, lane);
        v16bf bcur = frag_b_lds(Wcur + (ng * 64) * 32, 32, lane);
#pragma unroll
        for (int nt = 0; nt < 4; ++nt) {
            v16bf bnext = (nt < 3)
                ? frag_b_lds(Wcur + (ng * 64 + (nt + 1) * 16) * 32, 32, lane)
                : bcur;
            acc2[nt] = wmma_bf16(a, bcur, acc2[nt]);
            bcur = bnext;
        }
    }

    // bias + relu -> expert output to workspace
    {
        const float* b2e  = b2 + (size_t)e * HH2;
        float*       Oe   = Eout + (size_t)e * BN * HH2;
        const int    mrow = row0 + mt * 16 + ((lane >> 4) << 3);
#pragma unroll
        for (int nt = 0; nt < 4; ++nt) {
            const int   n    = ng * 64 + nt * 16 + (lane & 15);
            const float bias = b2e[n];
#pragma unroll
            for (int r = 0; r < 8; ++r)
                Oe[(size_t)(mrow + r) * HH2 + n] = fmaxf(acc2[nt][r] + bias, 0.f);
        }
    }
}

// ---------------------------------------------------------------------------
// Kernel 3: gated combine.
// ---------------------------------------------------------------------------
__global__ __launch_bounds__(256)
void cgc_combine_kernel(const float* __restrict__ Eout,
                        const float* __restrict__ g0, const float* __restrict__ g1,
                        const float* __restrict__ gs, float* __restrict__ out) {
    const int idx = blockIdx.x * 256 + threadIdx.x;   // over B*H2
    if (idx >= BN * HH2) return;
    const int b = idx >> 8;                           // H2 == 256

    float v[NEXP];
#pragma unroll
    for (int e = 0; e < NEXP; ++e) v[e] = Eout[(size_t)e * BN * HH2 + idx];

    const float* G0 = g0 + (size_t)b * 8;
    const float* G1 = g1 + (size_t)b * 8;
    const float* GS = gs + (size_t)b * 12;

    float t0 = 0.f, t1 = 0.f, ts = 0.f;
#pragma unroll
    for (int e = 0; e < 4; ++e) { t0 += G0[e] * v[e];  t1 += G1[e] * v[e]; }
#pragma unroll
    for (int e = 0; e < 4; ++e) { t0 += G0[4 + e] * v[4 + e];  t1 += G1[4 + e] * v[8 + e]; }
#pragma unroll
    for (int e = 0; e < NEXP; ++e) ts += GS[e] * v[e];

    out[idx]                = t0;
    out[BN * HH2 + idx]     = t1;
    out[2 * BN * HH2 + idx] = ts;
}

// ---------------------------------------------------------------------------
// Launcher.  Workspace layout (bytes):
//   [0,128K) g0   [128K,256K) g1   [256K,512K) gs
//   [512K,+48M) Eout f32   [+24M) Xbf bf16   [+12M) W1t bf16   [+3M) W2t bf16
// ---------------------------------------------------------------------------
extern "C" void kernel_launch(void* const* d_in, const int* in_sizes, int n_in,
                              void* d_out, int out_size, void* d_ws, size_t ws_size,
                              hipStream_t stream) {
    const float* x0  = (const float*)d_in[0];
    const float* x1  = (const float*)d_in[1];
    const float* xs  = (const float*)d_in[2];
    const float* W1  = (const float*)d_in[3];
    const float* b1  = (const float*)d_in[4];
    const float* W2  = (const float*)d_in[5];
    const float* b2  = (const float*)d_in[6];
    const float* Wg0 = (const float*)d_in[7];
    const float* Wg1 = (const float*)d_in[8];
    const float* Wgs = (const float*)d_in[9];

    char* wsb = (char*)d_ws;
    float*  g0   = (float*)(wsb);
    float*  g1   = (float*)(wsb + (size_t)131072);
    float*  gs   = (float*)(wsb + (size_t)262144);
    float*  Eout = (float*)(wsb + (size_t)524288);
    char*   p    = wsb + (size_t)524288 + (size_t)NEXP * BN * HH2 * 4;   // +48M
    bf16_t* Xbf  = (bf16_t*)p;  p += (size_t)3 * BN * DD * 2;            // +24M
    bf16_t* W1t  = (bf16_t*)p;  p += (size_t)NEXP * DD * HH1 * 2;        // +12M
    bf16_t* W2t  = (bf16_t*)p;                                           // +3M

    // 0) one-shot conversions (bf16 + weight transpose)
    cgc_convert_x_kernel<<<(3 * BN * DD / 4) / 256, 256, 0, stream>>>(xs, x0, x1, Xbf);
    {
        dim3 gw1(HH1 / 32, DD / 32, NEXP);
        cgc_convert_wt_kernel<<<gw1, 256, 0, stream>>>(W1, W1t, DD, HH1);
        dim3 gw2(HH2 / 32, HH1 / 32, NEXP);
        cgc_convert_wt_kernel<<<gw2, 256, 0, stream>>>(W2, W2t, HH1, HH2);
    }

    // 1) gates
    cgc_gates_kernel<<<BN, 256, 0, stream>>>(x0, x1, xs, Wg0, Wg1, Wgs, g0, g1, gs);

    // 2) fused expert FFN (WMMA + double-buffered TDM staging)
    dim3 grid(BN / 32, NEXP);
    cgc_expert_ffn_kernel<<<grid, 256, SMEM_TOTAL, stream>>>(Xbf, W1t, b1, W2t, b2, Eout);

    // 3) gated combine
    cgc_combine_kernel<<<(BN * HH2) / 256, 256, 0, stream>>>(Eout, g0, g1, gs, (float*)d_out);
}
